// Retention_14431090114834
// MI455X (gfx1250) — compile-verified, matrix operand order checked by reading, not run on hardware
//
#include <hip/hip_runtime.h>
#include <hip/hip_bf16.h>

// ---------------------------------------------------------------------------
// Retention forward for MI455X (gfx1250, wave32, WMMA).
//   Q = rot(x Wq^T + bq), K = rot(x Wk^T + bk), V = x Wv^T + bv
//   S = (Q K^T)/8 * exp(-sigmoid(gamma)*max(t-s,0));  O = S V
//   out = silu(x Wg^T + bg) * O   (heads re-interleaved)
// All GEMMs run on v_wmma_f32_16x16x32_f16 with f32 accumulation.
// ---------------------------------------------------------------------------

#define BB 2
#define TT 2048
#define EE 1024
#define HH 16
#define DD 64
#define MM (BB * TT)     // 4096 rows of x

typedef _Float16 v16h __attribute__((ext_vector_type(16)));
typedef _Float16 v8h  __attribute__((ext_vector_type(8)));
typedef float    v8f  __attribute__((ext_vector_type(8)));

static __device__ inline v8f wmma16x16x32(v16h a, v16h b, v8f c) {
    return __builtin_amdgcn_wmma_f32_16x16x32_f16(
        /*neg_a=*/false, a, /*neg_b=*/false, b,
        /*c_mod=*/(short)0, c, /*reuse_a=*/false, /*reuse_b=*/false);
}

static __device__ inline v8f zero8() {
    v8f z;
#pragma unroll
    for (int i = 0; i < 8; ++i) z[i] = 0.0f;
    return z;
}

// Load one 16-half fragment row slice for a lane.
// Hardware layout (ISA 7.12.2, 16-bit A/B 16x32): lane holds two contiguous
// 8-half chunks at k = hi*8 and k = hi*8 + 16.  `p` points at (row, hi*8).
static __device__ inline v16h load_frag(const _Float16* p) {
    v8h lo = *(const v8h*)(p);
    v8h hi = *(const v8h*)(p + 16);
    return __builtin_shufflevector(lo, hi, 0, 1, 2, 3, 4, 5, 6, 7,
                                           8, 9, 10, 11, 12, 13, 14, 15);
}

// ---------------------------------------------------------------------------
// f32 -> f16 conversion (x and the four weight matrices), grid-stride.
// ---------------------------------------------------------------------------
__global__ void cvt_f32_to_f16(const float* __restrict__ src,
                               _Float16* __restrict__ dst, int n) {
    int i = blockIdx.x * blockDim.x + threadIdx.x;
    int stride = gridDim.x * blockDim.x;
    for (; i < n; i += stride) dst[i] = (_Float16)src[i];
}

// ---------------------------------------------------------------------------
// Projection GEMM:  P[m,n] = sum_k xh[m,k] * W[n,k] + bias[n]
// blockIdx.z selects mode: 0=Q(rotary) 1=K(rotary) 2=V(store transposed)
// 3=gate(silu, f32).
// 128 threads = 4 waves (2x2), each wave a 32x64 tile (2x4 WMMA accumulators,
// 8 WMMAs per 12 b128 loads per k-step).
// ---------------------------------------------------------------------------
__global__ void proj_gemm(const _Float16* __restrict__ xh,
                          const _Float16* __restrict__ WhAll,
                          const float* __restrict__ bq,
                          const float* __restrict__ bk,
                          const float* __restrict__ bv,
                          const float* __restrict__ bg,
                          _Float16* __restrict__ Qh,
                          _Float16* __restrict__ Kh,
                          _Float16* __restrict__ Vt,
                          float* __restrict__ gate) {
    const int mode = blockIdx.z;
    const _Float16* W = WhAll + (size_t)mode * EE * EE;
    const float* bias = (mode == 0) ? bq : (mode == 1) ? bk
                      : (mode == 2) ? bv : bg;

    const int lane = threadIdx.x & 31;
    const int wave = threadIdx.x >> 5;          // 0..3
    const int lr   = lane & 15;
    const int hi8  = (lane >> 4) * 8;

    const int m0 = blockIdx.y * 64  + (wave >> 1) * 32;   // 2 row sub-tiles
    const int n0 = blockIdx.x * 128 + (wave & 1) * 64;    // 4 col sub-tiles

    v8f c[2][4];
#pragma unroll
    for (int i = 0; i < 2; ++i)
#pragma unroll
        for (int j = 0; j < 4; ++j) c[i][j] = zero8();

    const _Float16* arow = xh + (size_t)(m0 + lr) * EE + hi8;
    const _Float16* brow = W  + (size_t)(n0 + lr) * EE + hi8;

    for (int k = 0; k < EE; k += 32) {
        v16h a0 = load_frag(arow + k);
        v16h a1 = load_frag(arow + k + (size_t)16 * EE);
        v16h b0 = load_frag(brow + k);
        v16h b1 = load_frag(brow + k + (size_t)16 * EE);
        v16h b2 = load_frag(brow + k + (size_t)32 * EE);
        v16h b3 = load_frag(brow + k + (size_t)48 * EE);
        c[0][0] = wmma16x16x32(a0, b0, c[0][0]);
        c[0][1] = wmma16x16x32(a0, b1, c[0][1]);
        c[0][2] = wmma16x16x32(a0, b2, c[0][2]);
        c[0][3] = wmma16x16x32(a0, b3, c[0][3]);
        c[1][0] = wmma16x16x32(a1, b0, c[1][0]);
        c[1][1] = wmma16x16x32(a1, b1, c[1][1]);
        c[1][2] = wmma16x16x32(a1, b2, c[1][2]);
        c[1][3] = wmma16x16x32(a1, b3, c[1][3]);
    }

#pragma unroll
    for (int ti = 0; ti < 2; ++ti) {
#pragma unroll
        for (int ni = 0; ni < 4; ++ni) {
            const int n = n0 + ni * 16 + lr;     // output column (C layout: lane)
            const float bval = bias[n];
            const int mb = m0 + ti * 16 + hi8;
#pragma unroll
            for (int r = 0; r < 8; ++r) {
                const int mrow = mb + r;         // global row in (B*T)
                float val = c[ti][ni][r] + bval;
                if (mode <= 1) {
                    // rotary: partner element d^1 lives in lane^1 (same tile)
                    float part = __shfl_xor(val, 1, 32);
                    const int b = mrow >> 11;          // /T
                    const int t = mrow & (TT - 1);
                    const int h = n >> 6;              // /D
                    const int d = n & (DD - 1);
                    float xr = (d & 1) ? part : -part;
                    float inv_freq = __powf(10000.0f, -(float)(d & 31) * (1.0f / 32.0f));
                    float ang = (float)t * inv_freq;
                    float cs = __cosf(ang), sn = __sinf(ang);
                    float rv = val * cs + xr * sn;
                    _Float16* dst = (mode == 0) ? Qh : Kh;
                    dst[((size_t)(b * HH + h) * TT + t) * DD + d] = (_Float16)rv;
                } else if (mode == 2) {
                    const int b = mrow >> 11;
                    const int t = mrow & (TT - 1);
                    const int h = n >> 6;
                    const int d = n & (DD - 1);
                    // V stored transposed: (B,H,D,T) so attention B-frag loads
                    // are contiguous along s.
                    Vt[((size_t)(b * HH + h) * DD + d) * TT + t] = (_Float16)val;
                } else {
                    float s = val / (1.0f + __expf(-val));   // silu
                    gate[(size_t)mrow * EE + n] = s;
                }
            }
        }
    }
}

// ---------------------------------------------------------------------------
// Attention: one wave per 32-row t-tile of one (b,h); K/V fragments amortized
// over two 16-row t-subtiles (16 WMMAs per 16 global b128 loads per s-chunk).
// Decay handling is wave-uniform per 32-wide s-chunk:
//   s0 >  t0 : all diff <= 0 -> w = 1          (no exp)
//   s0 == t0 : diagonal chunk, per-element     (exp where diff > 0)
//   s0 <  t0 : all diff > 0  -> always exp; chunks with
//              g*(t0-s0-31) > 40 contribute < 4e-18 and are skipped.
// grid = (T/128, H, B), block = 128 (4 waves).
// ---------------------------------------------------------------------------
__global__ void attn_kernel(const _Float16* __restrict__ Qh,
                            const _Float16* __restrict__ Kh,
                            const _Float16* __restrict__ Vt,
                            const float* __restrict__ gate,
                            const float* __restrict__ gamma,
                            float* __restrict__ out) {
    const int b    = blockIdx.z;
    const int h    = blockIdx.y;
    const int lane = threadIdx.x & 31;
    const int wave = threadIdx.x >> 5;
    const int lr   = lane & 15;
    const int hi8  = (lane >> 4) * 8;

    const int t0 = (blockIdx.x * 4 + wave) * 32;

    const float g = 1.0f / (1.0f + __expf(-gamma[h]));   // sigmoid(gamma)

    const size_t bh = (size_t)(b * HH + h);
    const _Float16* Qbh = Qh + bh * TT * DD;
    const _Float16* Kbh = Kh + bh * TT * DD;
    const _Float16* Vbh = Vt + bh * DD * TT;

    // Q fragments (2 t-subtiles x 2 k-steps), registers for the whole s loop.
    v16h aq[2][2];
#pragma unroll
    for (int ts = 0; ts < 2; ++ts) {
        const _Float16* qp = Qbh + (size_t)(t0 + ts * 16 + lr) * DD + hi8;
        aq[ts][0] = load_frag(qp);
        aq[ts][1] = load_frag(qp + 32);
    }

    v8f o[2][4];
#pragma unroll
    for (int ts = 0; ts < 2; ++ts)
#pragma unroll
        for (int dt = 0; dt < 4; ++dt) o[ts][dt] = zero8();

    // Per-wave LDS tile: S_w, 32 (t) x 32 (s) halves, stride 40 halves so the
    // b128 fragment reads (t*80 + hi*16 bytes) stay 16-byte aligned.
    __shared__ _Float16 smem[4 * 32 * 40];
    _Float16* sw = smem + wave * 32 * 40;

    // Deep-past cutoff: skip chunks whose largest weight is < exp(-40).
    int s_start = 0;
    {
        int s_lim = t0 - 31 - (int)(40.0f / g);
        if (s_lim > 0) s_start = s_lim & ~31;
    }

    for (int s0 = s_start; s0 < TT; s0 += 32) {
        // ---- scores: S[t0..t0+31, s0..s0+31] ----
        v16h bk[2][2];
#pragma unroll
        for (int ss = 0; ss < 2; ++ss) {
            const _Float16* kp = Kbh + (size_t)(s0 + ss * 16 + lr) * DD + hi8;
            bk[ss][0] = load_frag(kp);
            bk[ss][1] = load_frag(kp + 32);
        }
        v8f s_[2][2];
#pragma unroll
        for (int ts = 0; ts < 2; ++ts)
#pragma unroll
            for (int ss = 0; ss < 2; ++ss) {
                v8f acc = zero8();
                acc = wmma16x16x32(aq[ts][0], bk[ss][0], acc);
                acc = wmma16x16x32(aq[ts][1], bk[ss][1], acc);
                s_[ts][ss] = acc;
            }

        // ---- decay weight + 1/sqrt(D), f32 -> f16 into LDS ----
        if (s0 > t0) {               // fully future: w = 1
#pragma unroll
            for (int ts = 0; ts < 2; ++ts)
#pragma unroll
                for (int ss = 0; ss < 2; ++ss)
#pragma unroll
                    for (int r = 0; r < 8; ++r)
                        sw[(ts * 16 + hi8 + r) * 40 + ss * 16 + lr] =
                            (_Float16)(s_[ts][ss][r] * 0.125f);
        } else if (s0 < t0) {        // fully past: always exp
#pragma unroll
            for (int ts = 0; ts < 2; ++ts)
#pragma unroll
                for (int ss = 0; ss < 2; ++ss)
#pragma unroll
                    for (int r = 0; r < 8; ++r) {
                        const int t = t0 + ts * 16 + hi8 + r;
                        const int s = s0 + ss * 16 + lr;
                        float v = s_[ts][ss][r] * 0.125f *
                                  __expf(-g * (float)(t - s));
                        sw[(ts * 16 + hi8 + r) * 40 + ss * 16 + lr] = (_Float16)v;
                    }
        } else {                     // diagonal chunk
#pragma unroll
            for (int ts = 0; ts < 2; ++ts)
#pragma unroll
                for (int ss = 0; ss < 2; ++ss)
#pragma unroll
                    for (int r = 0; r < 8; ++r) {
                        const int t = t0 + ts * 16 + hi8 + r;
                        const int s = s0 + ss * 16 + lr;
                        const int diff = t - s;
                        float v = s_[ts][ss][r] * 0.125f;
                        v *= (diff > 0) ? __expf(-g * (float)diff) : 1.0f;
                        sw[(ts * 16 + hi8 + r) * 40 + ss * 16 + lr] = (_Float16)v;
                    }
        }
        // Same-wave DS ops are in-order; compiler inserts s_wait_dscnt on the
        // register dependency before the WMMA consumes the fragment.

        // ---- re-read S_w as two 16x32 A-fragments ----
        v16h as0 = load_frag(sw + (size_t)lr * 40 + hi8);
        v16h as1 = load_frag(sw + (size_t)(16 + lr) * 40 + hi8);

        // ---- O[t, d] += S_w . V   (4 d-tiles, V-frag shared by t-subtiles) --
        const _Float16* vp = Vbh + (size_t)lr * TT + s0 + hi8;
#pragma unroll
        for (int dt = 0; dt < 4; ++dt) {
            v16h bv = load_frag(vp + (size_t)(dt * 16) * TT);
            o[0][dt] = wmma16x16x32(as0, bv, o[0][dt]);
            o[1][dt] = wmma16x16x32(as1, bv, o[1][dt]);
        }
    }

    // ---- gate * O, scatter to (B, T, H*D) fp32 ----
#pragma unroll
    for (int ts = 0; ts < 2; ++ts)
#pragma unroll
        for (int dt = 0; dt < 4; ++dt) {
            const int d = dt * 16 + lr;
            const int col = h * DD + d;
#pragma unroll
            for (int r = 0; r < 8; ++r) {
                const int t = t0 + ts * 16 + hi8 + r;
                const size_t idx = ((size_t)b * TT + t) * EE + col;
                out[idx] = gate[idx] * o[ts][dt][r];
            }
        }
}

// ---------------------------------------------------------------------------
// Host-side launch.
// Inputs: 0:x 1:Wq 2:bq 3:Wk 4:bk 5:Wv 6:bv 7:Wg 8:bg 9:gamma  (all f32)
// Workspace layout (56 MB):
//   [ 0MB) xh f16   [ 8MB) Wh f16 x4   [16MB) Qh   [24MB) Kh
//   [32MB) Vt       [40MB) gate f32
// ---------------------------------------------------------------------------
extern "C" void kernel_launch(void* const* d_in, const int* in_sizes, int n_in,
                              void* d_out, int out_size, void* d_ws, size_t ws_size,
                              hipStream_t stream) {
    (void)in_sizes; (void)n_in; (void)out_size; (void)ws_size;

    const float* x     = (const float*)d_in[0];
    const float* Wq    = (const float*)d_in[1];
    const float* bq    = (const float*)d_in[2];
    const float* Wk    = (const float*)d_in[3];
    const float* bk    = (const float*)d_in[4];
    const float* Wv    = (const float*)d_in[5];
    const float* bv    = (const float*)d_in[6];
    const float* Wg    = (const float*)d_in[7];
    const float* bg    = (const float*)d_in[8];
    const float* gamma = (const float*)d_in[9];

    const size_t MB = 1u << 20;
    char* ws = (char*)d_ws;
    _Float16* xh   = (_Float16*)(ws + 0 * MB);
    _Float16* Wh   = (_Float16*)(ws + 8 * MB);   // 4 matrices back to back
    _Float16* Qh   = (_Float16*)(ws + 16 * MB);
    _Float16* Kh   = (_Float16*)(ws + 24 * MB);
    _Float16* Vt   = (_Float16*)(ws + 32 * MB);
    float*    gate = (float*)   (ws + 40 * MB);

    // f32 -> f16 conversions
    {
        int n = MM * EE;
        cvt_f32_to_f16<<<dim3((n + 255) / 256), dim3(256), 0, stream>>>(x, xh, n);
        int nw = EE * EE;
        cvt_f32_to_f16<<<dim3((nw + 255) / 256), dim3(256), 0, stream>>>(Wq, Wh + 0 * (size_t)nw, nw);
        cvt_f32_to_f16<<<dim3((nw + 255) / 256), dim3(256), 0, stream>>>(Wk, Wh + 1 * (size_t)nw, nw);
        cvt_f32_to_f16<<<dim3((nw + 255) / 256), dim3(256), 0, stream>>>(Wv, Wh + 2 * (size_t)nw, nw);
        cvt_f32_to_f16<<<dim3((nw + 255) / 256), dim3(256), 0, stream>>>(Wg, Wh + 3 * (size_t)nw, nw);
    }

    // Four projections (z = Q, K, V, gate), WMMA GEMM.
    proj_gemm<<<dim3(EE / 128, MM / 64, 4), dim3(128), 0, stream>>>(
        xh, Wh, bq, bk, bv, bg, Qh, Kh, Vt, gate);

    // Attention + gating -> d_out (f32, (B,T,E)).
    attn_kernel<<<dim3(TT / 128, HH, BB), dim3(128), 0, stream>>>(
        Qh, Kh, Vt, gate, gamma, (float*)d_out);
}
